// AMCNN_32074815766597
// MI455X (gfx1250) — compile-verified, matrix operand order checked by reading, not run on hardware
//
#include <hip/hip_runtime.h>

typedef __attribute__((ext_vector_type(16))) __bf16 v16bf;
typedef __attribute__((ext_vector_type(8)))  float  v8f;

#define B_   128
#define L_   256
#define E_   300
#define H_   128
#define D_   256
#define C_   4
#define M_ROWS 32768   // B*L
#define KPAD0 320      // E padded to mult of 32
#define N3H  768       // 2 dirs * 3H

__device__ __forceinline__ v8f zero_v8() {
  v8f z;
  #pragma unroll
  for (int i = 0; i < 8; ++i) z[i] = 0.0f;
  return z;
}

// low 32 bits of a generic pointer to LDS == LDS byte address (ISA 10.2)
__device__ __forceinline__ unsigned lds_off(const void* p) {
  return (unsigned)(unsigned long long)p;
}

// async global->LDS copy of 16 bytes per lane (ASYNCcnt-tracked, bypasses VGPRs)
__device__ __forceinline__ void async_ld_b128(unsigned lds, unsigned long long gaddr) {
  asm volatile("global_load_async_to_lds_b128 %0, %1, off"
               :: "v"(lds), "v"(gaddr) : "memory");
}
__device__ __forceinline__ void wait_async0() {
  asm volatile("s_wait_asynccnt 0x0" ::: "memory");
}

// ---------------------------------------------------------------------------
// Generic tiled bf16 WMMA GEMM:  C[M x N] = A[M x K] * B[K x N]
// Block: 256 threads (8 waves), tile 128x64, K-step 32, double-buffered LDS
// filled with GLOBAL_LOAD_ASYNC_TO_LDS_B128.
// MODE 0: fp32 store. MODE 1: bf16 store. MODE 2: sigmoid(acc + bias[col]) fp32.
// Requires M%128==0, N%64==0, K%32==0; A/B base pointers 16B aligned,
// lda/ldb multiples of 8 (rows stay 16B aligned).
// ---------------------------------------------------------------------------
template <int MODE>
__global__ __launch_bounds__(256) void gemm_bf16_kernel(
    const __bf16* __restrict__ A, int lda, long strideA,
    const __bf16* __restrict__ Bm, int ldb, long strideB,
    void* __restrict__ Cp, int ldc, long strideC,
    const float* __restrict__ bias, int strideBias, int K) {
  const int batch = blockIdx.z;
  const __bf16* Ab = A + (long)batch * strideA;
  const __bf16* Bb = Bm + (long)batch * strideB;
  const float* biasb = (MODE == 2) ? (bias + (long)batch * strideBias) : nullptr;
  const int row0 = blockIdx.y * 128;
  const int col0 = blockIdx.x * 64;

  __shared__ __bf16 As[2][128 * 32];   // 2 x 8KB
  __shared__ __bf16 Bs[2][32 * 64];    // 2 x 4KB

  const int tid = threadIdx.x;
  const int lane = tid & 31;
  const int wave = tid >> 5;
  const int wm = wave >> 1;   // 0..3 -> 32-row band
  const int wn = wave & 1;    // 0..1 -> 32-col band
  const int m_lane = lane & 15;
  const int kb = (lane >> 4) << 3;   // 0 or 8

  // per-thread async chunk mapping (16B chunks)
  // A tile: 128 rows x 64B -> 512 chunks, 2 per thread
  const int cA0 = tid * 2;
  const int ar0 = cA0 >> 2, ab0 = (cA0 & 3) * 16;
  const int ar1 = (cA0 + 1) >> 2, ab1 = ((cA0 + 1) & 3) * 16;
  // B tile: 32 rows x 128B -> 256 chunks, 1 per thread
  const int br = tid >> 3, bbyte = (tid & 7) * 16;

  auto issue_slab = [&](int k0, int sel) {
    async_ld_b128(lds_off(&As[sel][0]) + (unsigned)(cA0 * 16),
                  (unsigned long long)(Ab + (long)(row0 + ar0) * lda + k0) + ab0);
    async_ld_b128(lds_off(&As[sel][0]) + (unsigned)((cA0 + 1) * 16),
                  (unsigned long long)(Ab + (long)(row0 + ar1) * lda + k0) + ab1);
    async_ld_b128(lds_off(&Bs[sel][0]) + (unsigned)(tid * 16),
                  (unsigned long long)(Bb + (long)(k0 + br) * ldb + col0) + bbyte);
  };

  v8f acc[2][2];
  #pragma unroll
  for (int i = 0; i < 2; ++i)
    #pragma unroll
    for (int j = 0; j < 2; ++j) acc[i][j] = zero_v8();

  const int nk = K / 32;
  issue_slab(0, 0);

  for (int s = 0; s < nk; ++s) {
    wait_async0();        // my lanes' slab-s chunks have landed
    __syncthreads();      // everyone's chunks have landed
    if (s + 1 < nk) issue_slab((s + 1) * 32, (s + 1) & 1);

    const __bf16* Asb = As[s & 1];
    const __bf16* Bsb = Bs[s & 1];

    v16bf af[2], bfr[2];
    #pragma unroll
    for (int ti = 0; ti < 2; ++ti) {
      int r = wm * 32 + ti * 16 + m_lane;
      #pragma unroll
      for (int v = 0; v < 8; ++v) {
        int k = (v < 4) ? (kb + 2 * v) : (16 + kb + 2 * (v - 4));
        af[ti][2 * v]     = Asb[r * 32 + k];
        af[ti][2 * v + 1] = Asb[r * 32 + k + 1];
      }
    }
    #pragma unroll
    for (int tj = 0; tj < 2; ++tj) {
      int cc = wn * 32 + tj * 16 + m_lane;
      #pragma unroll
      for (int v = 0; v < 8; ++v) {
        int k = (v < 4) ? (kb + 2 * v) : (16 + kb + 2 * (v - 4));
        bfr[tj][2 * v]     = Bsb[k * 64 + cc];
        bfr[tj][2 * v + 1] = Bsb[(k + 1) * 64 + cc];
      }
    }
    #pragma unroll
    for (int ti = 0; ti < 2; ++ti)
      #pragma unroll
      for (int tj = 0; tj < 2; ++tj)
        acc[ti][tj] = __builtin_amdgcn_wmma_f32_16x16x32_bf16(
            false, af[ti], false, bfr[tj], (short)0, acc[ti][tj], false, false);
    __syncthreads();      // all waves done reading slab s before it is overwritten
  }

  const int rsel = (lane >> 4) << 3;   // 0 or 8
  #pragma unroll
  for (int ti = 0; ti < 2; ++ti)
    #pragma unroll
    for (int tj = 0; tj < 2; ++tj) {
      int gc = col0 + wn * 32 + tj * 16 + m_lane;
      #pragma unroll
      for (int v = 0; v < 8; ++v) {
        int gr = row0 + wm * 32 + ti * 16 + v + rsel;
        float val = acc[ti][tj][v];
        if (MODE == 2) val = 1.0f / (1.0f + __expf(-(val + biasb[gc])));
        if (MODE == 1)
          ((__bf16*)Cp)[(long)batch * strideC + (long)gr * ldc + gc] = (__bf16)val;
        else
          ((float*)Cp)[(long)batch * strideC + (long)gr * ldc + gc] = val;
      }
    }
}

// ---------------------------------------------------------------------------
// Attention M kernel: per (c,b) compute s[i] = sum_j tanh( x[b,i,:].mat[c,b,j,:] + att_b[c] )
// WMMA with B = mat rows (transposed-K contraction); M is never materialized.
// j-slabs of mat[c,b] (16 rows x 512B = 8KB) are staged cooperatively into LDS
// with GLOBAL_LOAD_ASYNC_TO_LDS_B128, double-buffered, prefetching slab jt+1
// while slab jt feeds the WMMAs. A-fragments are hoisted in registers.
// ---------------------------------------------------------------------------
__global__ __launch_bounds__(256) void attn_m_kernel(
    const __bf16* __restrict__ x2b, const __bf16* __restrict__ matb,
    const float* __restrict__ att_b, float* __restrict__ slog) {
  __shared__ __bf16 Bsl[2][16 * 256];   // 2 x 8KB j-slabs (16 mat rows x K=256)
  const int cb = blockIdx.x;            // C*B
  const int c = cb >> 7, b = cb & 127;
  const int tid = threadIdx.x;
  const int lane = tid & 31;
  const int wave = tid >> 5;            // 0..7
  const int m_lane = lane & 15;
  const int kb = (lane >> 4) << 3;
  const __bf16* Abase = x2b + (long)b * L_ * D_;
  const __bf16* Bbase = matb + ((long)c * M_ROWS + (long)b * L_) * D_;
  const float bb = att_b[c];

  // slab staging: 512 chunks of 16B, 2 per thread (32 chunks per 512B row)
  const int ch0 = tid * 2;
  auto issue_slab = [&](int jt, int sel) {
    #pragma unroll
    for (int q = 0; q < 2; ++q) {
      int ch = ch0 + q;
      int rr = ch >> 5;
      int byt = (ch & 31) * 16;
      async_ld_b128(lds_off(&Bsl[sel][0]) + (unsigned)(ch * 16),
                    (unsigned long long)(Bbase + (long)(jt * 16 + rr) * D_) + byt);
    }
  };

  // hoist all A fragments for this wave's two i-tiles (rows wave*32 .. +31)
  v16bf af[2][8];
  #pragma unroll
  for (int ti = 0; ti < 2; ++ti) {
    const int r = wave * 32 + ti * 16 + m_lane;
    #pragma unroll
    for (int kc = 0; kc < 8; ++kc) {
      #pragma unroll
      for (int v = 0; v < 8; ++v) {
        int k = kc * 32 + ((v < 4) ? (kb + 2 * v) : (16 + kb + 2 * (v - 4)));
        af[ti][kc][2 * v]     = Abase[(long)r * D_ + k];
        af[ti][kc][2 * v + 1] = Abase[(long)r * D_ + k + 1];
      }
    }
  }

  float part[2][8];
  #pragma unroll
  for (int ti = 0; ti < 2; ++ti)
    #pragma unroll
    for (int v = 0; v < 8; ++v) part[ti][v] = 0.0f;

  issue_slab(0, 0);
  for (int jt = 0; jt < 16; ++jt) {
    wait_async0();
    __syncthreads();
    if (jt + 1 < 16) issue_slab(jt + 1, (jt + 1) & 1);

    const __bf16* Bsb = Bsl[jt & 1];
    v8f acc0 = zero_v8(), acc1 = zero_v8();
    #pragma unroll
    for (int kc = 0; kc < 8; ++kc) {
      v16bf bfr;   // shared by both i-tiles: lane n = mat local row n
      #pragma unroll
      for (int v = 0; v < 8; ++v) {
        int k = kc * 32 + ((v < 4) ? (kb + 2 * v) : (16 + kb + 2 * (v - 4)));
        bfr[2 * v]     = Bsb[m_lane * 256 + k];
        bfr[2 * v + 1] = Bsb[m_lane * 256 + k + 1];
      }
      acc0 = __builtin_amdgcn_wmma_f32_16x16x32_bf16(
          false, af[0][kc], false, bfr, (short)0, acc0, false, false);
      acc1 = __builtin_amdgcn_wmma_f32_16x16x32_bf16(
          false, af[1][kc], false, bfr, (short)0, acc1, false, false);
    }
    #pragma unroll
    for (int v = 0; v < 8; ++v) {
      part[0][v] += tanhf(acc0[v] + bb);
      part[1][v] += tanhf(acc1[v] + bb);
    }
    __syncthreads();   // done reading slab jt before it is overwritten
  }

  // reduce over the 16 N-lanes inside each half-wave, store s[i]
  #pragma unroll
  for (int ti = 0; ti < 2; ++ti) {
    #pragma unroll
    for (int v = 0; v < 8; ++v) {
      float s = part[ti][v];
      for (int m = 1; m < 16; m <<= 1) s += __shfl_xor(s, m, 32);
      if (m_lane == 0) {
        int i = wave * 32 + ti * 16 + v + ((lane >= 16) ? 8 : 0);
        slog[((long)c * B_ + b) * L_ + i] = s;
      }
    }
  }
}

// ------------------------- elementwise / small kernels ----------------------
__global__ void embed_kernel(const int* __restrict__ utt,
                             const float* __restrict__ emb,
                             __bf16* __restrict__ x0b) {
  long gid = (long)blockIdx.x * blockDim.x + threadIdx.x;
  if (gid >= (long)M_ROWS * KPAD0) return;
  int e = (int)(gid % KPAD0);
  long row = gid / KPAD0;
  float v = 0.0f;
  if (e < E_) {
    int tok = utt[row];
    v = emb[(long)tok * E_ + e];
  }
  x0b[gid] = (__bf16)v;
}

__global__ void cast_f2b_kernel(const float* __restrict__ in,
                                __bf16* __restrict__ out, long n) {
  long i = (long)blockIdx.x * blockDim.x + threadIdx.x;
  if (i < n) out[i] = (__bf16)in[i];
}

__global__ void zero_b_kernel(__bf16* __restrict__ p, long n) {
  long i = (long)blockIdx.x * blockDim.x + threadIdx.x;
  if (i < n) p[i] = (__bf16)0.0f;
}

__global__ void pack_wih_kernel(const float* __restrict__ w_ih, int Kin, int Kpad,
                                __bf16* __restrict__ out) {
  long gid = (long)blockIdx.x * blockDim.x + threadIdx.x;
  if (gid >= (long)Kpad * N3H) return;
  int col = (int)(gid % N3H);
  long k = gid / N3H;
  int dir = col / 384, gj = col % 384;
  float v = (k < Kin) ? w_ih[((long)dir * 384 + gj) * Kin + k] : 0.0f;
  out[gid] = (__bf16)v;
}

__global__ void pack_conv_kernel(const float* __restrict__ cw, int fs, int off,
                                 __bf16* __restrict__ Wcb) {
  long total = (long)fs * C_ * fs * D_;
  long gid = (long)blockIdx.x * blockDim.x + threadIdx.x;
  if (gid >= total) return;
  int w = (int)(gid % D_);
  long r = gid / D_;
  int dh = (int)(r % fs); r /= fs;
  int c = (int)(r % C_);  r /= C_;
  int o = (int)r;
  float v = cw[(((long)o * C_ + c) * fs + dh) * D_ + w];
  Wcb[(long)w * 256 + off + ((o * C_ + c) * fs + dh)] = (__bf16)v;
}

// GRU recurrence: one block per direction, 1024 threads, h state in LDS.
__global__ __launch_bounds__(1024) void gru_scan_kernel(
    const float* __restrict__ gx, const float* __restrict__ w_hh,
    const float* __restrict__ b_ih, const float* __restrict__ b_hh,
    float* __restrict__ xout) {
  __shared__ float hs[B_ * H_];
  const int dir = blockIdx.x;
  const int tid = threadIdx.x;
  const int j = tid & 127;
  const int bq = tid >> 7;     // 0..7
  const float* w  = w_hh + (long)dir * 384 * H_;
  const float* bi = b_ih + dir * 384;
  const float* bh = b_hh + dir * 384;
  for (int e = tid; e < B_ * H_; e += 1024) hs[e] = 0.0f;
  __syncthreads();
  float hnew[16];
  const float* wr = w + (0 * H_ + j) * H_;
  const float* wz = w + ((H_ + j)) * H_;
  const float* wn = w + ((2 * H_ + j)) * H_;
  for (int t = 0; t < L_; ++t) {
    const int tt = dir ? (L_ - 1 - t) : t;
    for (int r = 0; r < 16; ++r) {
      const int b = bq * 16 + r;
      const float* hrow = hs + b * H_;
      float dr = 0.f, dz = 0.f, dn = 0.f;
      for (int k = 0; k < H_; ++k) {
        float hv = hrow[k];
        dr += hv * wr[k]; dz += hv * wz[k]; dn += hv * wn[k];
      }
      const float* gxr = gx + ((long)b * L_ + tt) * N3H + (long)dir * 384;
      float xr = gxr[j] + bi[j];
      float xz = gxr[H_ + j] + bi[H_ + j];
      float xn = gxr[2 * H_ + j] + bi[2 * H_ + j];
      float rr = 1.0f / (1.0f + __expf(-(xr + dr + bh[j])));
      float zz = 1.0f / (1.0f + __expf(-(xz + dz + bh[H_ + j])));
      float nn = tanhf(xn + rr * (dn + bh[2 * H_ + j]));
      hnew[r] = (1.0f - zz) * nn + zz * hrow[j];
    }
    __syncthreads();
    for (int r = 0; r < 16; ++r) {
      const int b = bq * 16 + r;
      hs[b * H_ + j] = hnew[r];
      xout[((long)b * L_ + tt) * D_ + dir * H_ + j] = hnew[r];
    }
    __syncthreads();
  }
}

__global__ __launch_bounds__(256) void softmax256_kernel(float* __restrict__ x) {
  __shared__ float red[256];
  const long row = blockIdx.x;
  const int t = threadIdx.x;
  float v = x[row * 256 + t];
  red[t] = v; __syncthreads();
  for (int s = 128; s > 0; s >>= 1) { if (t < s) red[t] = fmaxf(red[t], red[t + s]); __syncthreads(); }
  float mx = red[0]; __syncthreads();
  float e = __expf(v - mx);
  red[t] = e; __syncthreads();
  for (int s = 128; s > 0; s >>= 1) { if (t < s) red[t] += red[t + s]; __syncthreads(); }
  x[row * 256 + t] = e / red[0];
}

__global__ __launch_bounds__(256) void aalogit_kernel(
    const float* __restrict__ g, const float* __restrict__ wv1,
    float* __restrict__ aalog) {
  const int cb = blockIdx.x;
  const int c = cb >> 7, b = cb & 127;
  const int l = threadIdx.x;
  const float* grow = g + ((long)c * M_ROWS + (long)b * L_ + l) * D_;
  const float* wv = wv1 + c * D_;
  float s = 0.0f;
  for (int e = 0; e < D_; ++e) s += grow[e] * wv[e];
  aalog[((long)c * B_ + b) * L_ + l] = s;
}

__global__ __launch_bounds__(256) void newv_kernel(
    const float* __restrict__ aa, const float* __restrict__ x2,
    float* __restrict__ newf, __bf16* __restrict__ newb) {
  const int cb = blockIdx.x;
  const int c = cb >> 7, b = cb & 127;
  const int d = threadIdx.x;
  const float* arow = aa + ((long)c * B_ + b) * L_;
  float s = 0.0f;
  for (int l = 0; l < L_; ++l) s += arow[l] * x2[((long)b * L_ + l) * D_ + d];
  long o = ((long)c * B_ + b) * D_ + d;
  newf[o] = s;
  newb[o] = (__bf16)s;
}

// Conv (collapsed) + ReLU + MaxPool + FC
__global__ __launch_bounds__(256) void conv_fc_kernel(
    const float* __restrict__ a, const float* __restrict__ P,
    const float* __restrict__ Rq,
    const float* __restrict__ cb0, const float* __restrict__ cb1,
    const float* __restrict__ cb2, const float* __restrict__ cb3,
    const float* __restrict__ fc_w, const float* __restrict__ fc_b,
    float* __restrict__ out) {
  __shared__ float red[256];
  __shared__ float feats[14];
  const int b = blockIdx.x;
  const int i = threadIdx.x;
  const int FSs[4]  = {2, 3, 4, 5};
  const int offs[4] = {0, 16, 52, 116};
  const float* cbs[4] = {cb0, cb1, cb2, cb3};
  int of = 0;
  for (int fi = 0; fi < 4; ++fi) {
    const int fs = FSs[fi];
    for (int o = 0; o < fs; ++o) {
      float val = -1e30f;
      if (i <= L_ - fs) {
        float s = cbs[fi][o];
        for (int c = 0; c < C_; ++c)
          for (int dh = 0; dh < fs; ++dh) {
            int jc = offs[fi] + ((o * C_ + c) * fs + dh);
            s += a[((long)c * B_ + b) * L_ + (i + dh)] *
                     P[((long)b * L_ + i + dh) * 256 + jc] +
                 Rq[((long)c * B_ + b) * 256 + jc];
          }
        val = fmaxf(s, 0.0f);
      }
      red[i] = val; __syncthreads();
      for (int st = 128; st > 0; st >>= 1) { if (i < st) red[i] = fmaxf(red[i], red[i + st]); __syncthreads(); }
      if (i == 0) feats[of] = red[0];
      __syncthreads();
      ++of;
    }
  }
  if (i < 2) {
    float s = fc_b[i];
    for (int f = 0; f < 14; ++f) s += feats[f] * fc_w[i * 14 + f];
    out[(long)b * 2 + i] = s;
  }
}

// ---------------------------------------------------------------------------
extern "C" void kernel_launch(void* const* d_in, const int* in_sizes, int n_in,
                              void* d_out, int out_size, void* d_ws, size_t ws_size,
                              hipStream_t stream) {
  (void)in_sizes; (void)n_in; (void)out_size; (void)ws_size;
  const int*   utt    = (const int*)d_in[0];
  const float* emb    = (const float*)d_in[2];
  const float* w_ih0  = (const float*)d_in[3];
  const float* w_hh0  = (const float*)d_in[4];
  const float* b_ih0  = (const float*)d_in[5];
  const float* b_hh0  = (const float*)d_in[6];
  const float* w_ih1  = (const float*)d_in[7];
  const float* w_hh1  = (const float*)d_in[8];
  const float* b_ih1  = (const float*)d_in[9];
  const float* b_hh1  = (const float*)d_in[10];
  const float* att_w  = (const float*)d_in[11];
  const float* att_b  = (const float*)d_in[12];
  const float* att_wv2= (const float*)d_in[13];
  const float* att_bv = (const float*)d_in[14];
  const float* att_wv1= (const float*)d_in[15];
  const float* fc_w   = (const float*)d_in[16];
  const float* fc_b   = (const float*)d_in[17];
  const float* cw[4]  = {(const float*)d_in[18], (const float*)d_in[20],
                         (const float*)d_in[22], (const float*)d_in[24]};
  const float* cbv[4] = {(const float*)d_in[19], (const float*)d_in[21],
                         (const float*)d_in[23], (const float*)d_in[25]};
  float* outp = (float*)d_out;

  char* ws = (char*)d_ws;
  size_t off = 0;
  auto alloc = [&](size_t bytes) { size_t o = off; off += (bytes + 255) & ~(size_t)255; return o; };
  size_t BIG  = alloc((size_t)C_ * M_ROWS * D_ * 4);      // gx0 -> gx1 -> g (134 MB)
  size_t X0B  = alloc((size_t)M_ROWS * KPAD0 * 2);
  size_t X1   = alloc((size_t)M_ROWS * D_ * 4);
  size_t X1B  = alloc((size_t)M_ROWS * D_ * 2);
  size_t X2   = alloc((size_t)M_ROWS * D_ * 4);
  size_t X2B  = alloc((size_t)M_ROWS * D_ * 2);
  size_t MATB = alloc((size_t)C_ * M_ROWS * D_ * 2);
  size_t WIH0 = alloc((size_t)KPAD0 * N3H * 2);
  size_t WIH1 = alloc((size_t)D_ * N3H * 2);
  size_t ATTW = alloc((size_t)C_ * D_ * D_ * 2);
  size_t ATW2 = alloc((size_t)C_ * D_ * D_ * 2);
  size_t WCB  = alloc((size_t)D_ * 256 * 2);
  size_t SLOG = alloc((size_t)C_ * B_ * L_ * 4);
  size_t AAL  = alloc((size_t)C_ * B_ * L_ * 4);
  size_t NEWF = alloc((size_t)C_ * B_ * D_ * 4);
  size_t NEWB = alloc((size_t)C_ * B_ * D_ * 2);
  size_t PBUF = alloc((size_t)M_ROWS * 256 * 4);
  size_t RQ   = alloc((size_t)C_ * B_ * 256 * 4);

  __bf16* x0b   = (__bf16*)(ws + X0B);
  float*  big   = (float*)(ws + BIG);
  float*  x1    = (float*)(ws + X1);
  __bf16* x1b   = (__bf16*)(ws + X1B);
  float*  x2    = (float*)(ws + X2);
  __bf16* x2b   = (__bf16*)(ws + X2B);
  __bf16* matb  = (__bf16*)(ws + MATB);
  __bf16* wih0b = (__bf16*)(ws + WIH0);
  __bf16* wih1b = (__bf16*)(ws + WIH1);
  __bf16* attwb = (__bf16*)(ws + ATTW);
  __bf16* atw2b = (__bf16*)(ws + ATW2);
  __bf16* wcb   = (__bf16*)(ws + WCB);
  float*  slog  = (float*)(ws + SLOG);
  float*  aal   = (float*)(ws + AAL);
  float*  newf  = (float*)(ws + NEWF);
  __bf16* newb  = (__bf16*)(ws + NEWB);
  float*  pbuf  = (float*)(ws + PBUF);
  float*  rq    = (float*)(ws + RQ);

  const int convFS[4]  = {2, 3, 4, 5};
  const int convOff[4] = {0, 16, 52, 116};

  // ---- weight packing / casts ----
  pack_wih_kernel<<<(KPAD0 * N3H + 255) / 256, 256, 0, stream>>>(w_ih0, E_, KPAD0, wih0b);
  pack_wih_kernel<<<(D_ * N3H + 255) / 256, 256, 0, stream>>>(w_ih1, D_, D_, wih1b);
  cast_f2b_kernel<<<(C_ * D_ * D_ + 255) / 256, 256, 0, stream>>>(att_w, attwb, (long)C_ * D_ * D_);
  cast_f2b_kernel<<<(C_ * D_ * D_ + 255) / 256, 256, 0, stream>>>(att_wv2, atw2b, (long)C_ * D_ * D_);
  zero_b_kernel<<<(D_ * 256 + 255) / 256, 256, 0, stream>>>(wcb, (long)D_ * 256);
  for (int fi = 0; fi < 4; ++fi) {
    long n = (long)convFS[fi] * C_ * convFS[fi] * D_;
    pack_conv_kernel<<<(int)((n + 255) / 256), 256, 0, stream>>>(cw[fi], convFS[fi], convOff[fi], wcb);
  }
  embed_kernel<<<(int)(((long)M_ROWS * KPAD0 + 255) / 256), 256, 0, stream>>>(utt, emb, x0b);

  // ---- GRU layer 0 ----
  gemm_bf16_kernel<0><<<dim3(N3H / 64, M_ROWS / 128, 1), 256, 0, stream>>>(
      x0b, KPAD0, 0, wih0b, N3H, 0, big, N3H, 0, nullptr, 0, KPAD0);
  gru_scan_kernel<<<2, 1024, 0, stream>>>(big, w_hh0, b_ih0, b_hh0, x1);
  cast_f2b_kernel<<<(int)(((long)M_ROWS * D_ + 255) / 256), 256, 0, stream>>>(x1, x1b, (long)M_ROWS * D_);

  // ---- GRU layer 1 ----
  gemm_bf16_kernel<0><<<dim3(N3H / 64, M_ROWS / 128, 1), 256, 0, stream>>>(
      x1b, D_, 0, wih1b, N3H, 0, big, N3H, 0, nullptr, 0, D_);
  gru_scan_kernel<<<2, 1024, 0, stream>>>(big, w_hh1, b_ih1, b_hh1, x2);
  cast_f2b_kernel<<<(int)(((long)M_ROWS * D_ + 255) / 256), 256, 0, stream>>>(x2, x2b, (long)M_ROWS * D_);

  // ---- attention: mat (bf16 out), M+tanh+rowsum, softmax -> a ----
  gemm_bf16_kernel<1><<<dim3(D_ / 64, M_ROWS / 128, C_), 256, 0, stream>>>(
      x2b, D_, 0, attwb, D_, (long)D_ * D_, matb, D_, (long)M_ROWS * D_, nullptr, 0, D_);
  attn_m_kernel<<<C_ * B_, 256, 0, stream>>>(x2b, matb, att_b, slog);
  softmax256_kernel<<<C_ * B_, 256, 0, stream>>>(slog);

  // ---- g = sigmoid(x@wv2 + bv), aa logits, softmax -> aa, new ----
  gemm_bf16_kernel<2><<<dim3(D_ / 64, M_ROWS / 128, C_), 256, 0, stream>>>(
      x2b, D_, 0, atw2b, D_, (long)D_ * D_, big, D_, (long)M_ROWS * D_, att_bv, D_, D_);
  aalogit_kernel<<<C_ * B_, 256, 0, stream>>>(big, att_wv1, aal);
  softmax256_kernel<<<C_ * B_, 256, 0, stream>>>(aal);
  newv_kernel<<<C_ * B_, 256, 0, stream>>>(aal, x2, newf, newb);

  // ---- conv as GEMM: P = x2 @ Wc, Rq = new @ Wc ----
  gemm_bf16_kernel<0><<<dim3(256 / 64, M_ROWS / 128, 1), 256, 0, stream>>>(
      x2b, D_, 0, wcb, 256, 0, pbuf, 256, 0, nullptr, 0, D_);
  gemm_bf16_kernel<0><<<dim3(256 / 64, (C_ * B_) / 128, 1), 256, 0, stream>>>(
      newb, D_, 0, wcb, 256, 0, rq, 256, 0, nullptr, 0, D_);

  // ---- pool + fc ----
  conv_fc_kernel<<<B_, 256, 0, stream>>>(slog, pbuf, rq, cbv[0], cbv[1], cbv[2], cbv[3],
                                         fc_w, fc_b, outp);
}
